// ConfidenceAwareSSIM_35244501631148
// MI455X (gfx1250) — matrix operand order repaired and checked by reading, not compiled, not run on hardware
//
#include <hip/hip_runtime.h>
#include <hip/hip_bf16.h>

typedef _Float16 v16h __attribute__((ext_vector_type(16)));
typedef float    v8f  __attribute__((ext_vector_type(8)));

#define H_IMG 512
#define W_IMG 512
#define BATCH 16
#define CHAN  3
#define TILE  16
#define HALO  26      // TILE + 2*5 halo
#define HSTR  33      // LDS halo row stride: 32 data cols (26 real + 6 zero pad) + 1
#define HBSTR 17      // LDS H-buffer row stride

// CDNA5 WMMA f16 A-matrix (16x32) element->K mapping for element j of lane-half hi
__device__ __forceinline__ int kA(int j, int hi) {
    int vg = j >> 1;
    return ((vg & 4) ? 16 : 0) + 8 * hi + 2 * (vg & 3) + (j & 1);
}
// CDNA5 WMMA f16 B-matrix (32x16) element->K mapping
__device__ __forceinline__ int kB(int j, int hi) {
    return kA(j, hi);
}

// One separable 11-tap conv of a 16x16 tile using 3 WMMAs:
//   H(26x16) = Halo(26x32) x Wh(32x16)   [2 WMMAs: row blocks 0..15 and 10..25]
//   Out(16x16) = Wv(16x32) x H(32x16)    [1 WMMA]
// Double-buffered caller (alternating hb) removes the WAR barrier: only the
// RAW (store -> gather) ordering point remains.
__device__ __forceinline__ v8f conv_field(v16h a0, v16h a1, v16h wh, v16h wv,
                                          float (*hb)[HBSTR], int n, int hi) {
    v8f z = {};
    v8f c0 = __builtin_amdgcn_wmma_f32_16x16x32_f16(false, a0, false, wh, (short)0, z, false, false);
    v8f c1 = __builtin_amdgcn_wmma_f32_16x16x32_f16(false, a1, false, wh, (short)0, z, false, false);
#pragma unroll
    for (int v = 0; v < 8; ++v) hb[v + 8 * hi][n] = c0[v];          // H rows 0..15
#pragma unroll
    for (int v = 0; v < 8; ++v) hb[10 + v + 8 * hi][n] = c1[v];     // H rows 10..25
    __syncthreads();   // single-wave WG -> hw S_NOP; orders LDS store->gather
    v16h b;
#pragma unroll
    for (int j = 0; j < 16; ++j) b[j] = (_Float16)hb[kB(j, hi)][n];
    return __builtin_amdgcn_wmma_f32_16x16x32_f16(false, wv, false, b, (short)0, z, false, false);
}

__global__ void ConfidenceAwareSSIM_kernel(const float* __restrict__ xg,
                                           const float* __restrict__ yg,
                                           const float* __restrict__ cg,
                                           float* __restrict__ out) {
    __shared__ float HX[HALO][HSTR];
    __shared__ float HY[HALO][HSTR];
    __shared__ float HB[2][32][HBSTR];   // ping-pong H-buffers
    __shared__ float WTAB[64];           // zero-padded 1D Gaussian, centered at +16

    const int lane = threadIdx.x;
    const int tilesX = W_IMG / TILE, tilesY = H_IMG / TILE;   // 32, 32 (pow2)
    int t  = blockIdx.x;
    const int tx = t & (tilesX - 1);  t >>= 5;
    const int ty = t & (tilesY - 1);  t >>= 5;
    const int bc = t;                       // b*CHAN + c, 0..47
    const int x0 = tx * TILE, y0 = ty * TILE;

    const float* xp = xg + (size_t)bc * (H_IMG * W_IMG);
    const float* yp = yg + (size_t)bc * (H_IMG * W_IMG);
    const float* cp = cg + (size_t)(bc / CHAN) * (H_IMG * W_IMG);

    // --- Zero-padded weight table (branchless: ternary has no memory op).
    // g[d] = exp(-(d-5)^2 / (2*1.5^2)) / sum,  1/sum = 0.26601180f
#pragma unroll
    for (int i = lane; i < 64; i += 32) {
        int   d  = i - 16;
        float dd = (float)(d - 5);
        float w  = __expf(-dd * dd * (1.0f / 4.5f)) * 0.26601180f;
        WTAB[i]  = (d >= 0 && d <= 10) ? w : 0.0f;
    }
    // --- Zero K-padding: halo cols 26..31 and H-buffer rows 26..31 (both buffers).
    if (lane < HALO) {
#pragma unroll
        for (int c = 0; c < 6; ++c) { HX[lane][26 + c] = 0.0f; HY[lane][26 + c] = 0.0f; }
    }
#pragma unroll
    for (int f = 0; f < 2; ++f)
        for (int i = lane; i < 6 * 16; i += 32)
            HB[f][26 + (i >> 4)][i & 15] = 0.0f;

    // --- Async halo load (CDNA5 global->LDS DMA path), addresses clamped
    //     in-range so the loop is branch-free; edges fixed up below.
    const uint32_t ldsX0 = (uint32_t)(uintptr_t)(&HX[0][0]);
    const uint32_t ldsY0 = (uint32_t)(uintptr_t)(&HY[0][0]);
    for (int i = lane; i < HALO * HALO; i += 32) {
        int hr = i / HALO, hc = i - hr * HALO;
        int gy = y0 - 5 + hr, gx = x0 - 5 + hc;
        int cy = min(max(gy, 0), H_IMG - 1);
        int cx = min(max(gx, 0), W_IMG - 1);
        size_t   gi  = (size_t)cy * W_IMG + cx;
        uint32_t off = (uint32_t)((hr * HSTR + hc) * 4);
        asm volatile("global_load_async_to_lds_b32 %0, %1, off"
                     :: "v"(ldsX0 + off), "v"(xp + gi) : "memory");
        asm volatile("global_load_async_to_lds_b32 %0, %1, off"
                     :: "v"(ldsY0 + off), "v"(yp + gi) : "memory");
    }
    asm volatile("s_wait_asynccnt 0x0" ::: "memory");

    // --- Boundary tiles only: overwrite OOB halo slots with zeros (zero-pad conv).
    if ((tx == 0) | (tx == tilesX - 1) | (ty == 0) | (ty == tilesY - 1)) {
        for (int i = lane; i < HALO * HALO; i += 32) {
            int hr = i / HALO, hc = i - hr * HALO;
            int gy = y0 - 5 + hr, gx = x0 - 5 + hc;
            if (((unsigned)gy >= (unsigned)H_IMG) | ((unsigned)gx >= (unsigned)W_IMG)) {
                HX[hr][hc] = 0.0f;
                HY[hr][hc] = 0.0f;
            }
        }
    }
    __syncthreads();

    const int m = lane & 15, hi = lane >> 4, n = m;

    // --- Constant banded fragments, unconditional LDS gathers.
    v16h wh, wv;
#pragma unroll
    for (int j = 0; j < 16; ++j) {
        wh[j] = (_Float16)WTAB[16 + kB(j, hi) - m];
        wv[j] = (_Float16)WTAB[16 + kA(j, hi) - m];
    }

    // --- Halo A-fragments (row blocks 0..15 and 10..25), unconditional gathers.
    v16h ax0, ax1, ay0, ay1;
#pragma unroll
    for (int j = 0; j < 16; ++j) {
        int k = kA(j, hi);
        ax0[j] = (_Float16)HX[m][k];
        ay0[j] = (_Float16)HY[m][k];
        ax1[j] = (_Float16)HX[m + 10][k];
        ay1[j] = (_Float16)HY[m + 10][k];
    }

    // --- 5 filtered fields (15 WMMAs), ping-pong H-buffers (WAR barrier-free).
    v8f mux = conv_field(ax0,       ax1,       wh, wv, HB[0], n, hi);
    v8f muy = conv_field(ay0,       ay1,       wh, wv, HB[1], n, hi);
    v8f exx = conv_field(ax0 * ax0, ax1 * ax1, wh, wv, HB[0], n, hi);  // v_pk_mul_f16
    v8f eyy = conv_field(ay0 * ay0, ay1 * ay1, wh, wv, HB[1], n, hi);
    v8f exy = conv_field(ax0 * ay0, ax1 * ay1, wh, wv, HB[0], n, hi);

    // --- Pointwise SSIM in C-layout: element v -> (row y0+v+8*hi, col x0+m).
    const float C1 = 1e-4f, C2 = 9e-4f;
    float s = 0.0f;
#pragma unroll
    for (int v = 0; v < 8; ++v) {
        int row = y0 + v + 8 * hi, col = x0 + m;
        float cf  = cp[(size_t)row * W_IMG + col];
        float mx  = mux[v], my = muy[v];
        float mx2 = mx * mx, my2 = my * my, mxy = mx * my;
        float sx2 = exx[v] - mx2, sy2 = eyy[v] - my2, sxy = exy[v] - mxy;
        float num = (2.0f * mxy + C1) * (2.0f * sxy + C2);
        float den = (mx2 + my2 + C1) * (sx2 + sy2 + C2);
        float loss = 1.0f - num / den;
        loss = fminf(fmaxf(loss, 0.0f), 1.0f);
        s += loss * cf;
    }
    // --- wave32 butterfly reduction, then one atomic per tile.
#pragma unroll
    for (int off = 16; off; off >>= 1) s += __shfl_xor(s, off, 32);
    if (lane == 0)
        atomicAdd(out, s * (1.0f / (float)((size_t)BATCH * CHAN * H_IMG * W_IMG)));
}

extern "C" void kernel_launch(void* const* d_in, const int* in_sizes, int n_in,
                              void* d_out, int out_size, void* d_ws, size_t ws_size,
                              hipStream_t stream) {
    (void)in_sizes; (void)n_in; (void)out_size; (void)d_ws; (void)ws_size;
    const float* x    = (const float*)d_in[0];
    const float* y    = (const float*)d_in[1];
    const float* conf = (const float*)d_in[2];
    float* out = (float*)d_out;

    hipMemsetAsync(out, 0, sizeof(float), stream);   // capture-safe zeroing

    const int tiles = BATCH * CHAN * (H_IMG / TILE) * (W_IMG / TILE);  // 49152
    ConfidenceAwareSSIM_kernel<<<tiles, 32, 0, stream>>>(x, y, conf, out);
}
